// ECGattCGNN_61014305407141
// MI455X (gfx1250) — matrix operand order repaired and checked by reading, not compiled, not run on hardware
//
#include <hip/hip_runtime.h>
#include <hip/hip_bf16.h>
#include <math.h>

// ---------- WMMA types & helpers (gfx1250, wave32) ----------
typedef __attribute__((ext_vector_type(16))) _Float16 v16h;
typedef __attribute__((ext_vector_type(8)))  float    v8f;
typedef __attribute__((ext_vector_type(4)))  unsigned int v4u;
typedef __attribute__((ext_vector_type(8)))  int      v8i;
typedef __attribute__((ext_vector_type(4)))  int      v4i;

union Frag16 { v16h v; _Float16 h[16]; };

__device__ __forceinline__ v8f wmma_f16(v16h a, v16h b, v8f c) {
    // D = A(16x32 f16) x B(32x16 f16) + C(16x16 f32)
    return __builtin_amdgcn_wmma_f32_16x16x32_f16(
        /*neg_a=*/false, a, /*neg_b=*/false, b,
        /*c_mod=*/(short)0, c, /*reuse_a=*/false, /*reuse_b=*/false);
}

// TDM: contiguous 1-row tensor load (nelem 4-byte elements) global -> LDS.
// D# built per CDNA5 ISA 8.3-8.6 (group0: flags/lds/global/type, group1: dims).
// This toolchain exposes the 6-arg builtin: (g0 v4u, g1 v8i, g2 v4i, g3 v4i,
// g4 v8i, cpol i32).
__device__ __forceinline__ void tdm_load_1d_f32(unsigned int lds_off,
                                                const void* gptr,
                                                unsigned int nelem) {
    unsigned long long ga = (unsigned long long)(uintptr_t)gptr;
    v4u g0;
    g0[0] = 1u;                                   // count=1, user mode
    g0[1] = lds_off;                              // lds_addr
    g0[2] = (unsigned int)(ga & 0xFFFFFFFFu);     // global_addr[31:0]
    g0[3] = (unsigned int)((ga >> 32) & 0x01FFFFFFu) | 0x80000000u; // [56:32], type=2
    v8i g1;
    g1[0] = 0x00020000;                           // wg_mask=0, data_size=2 (4B)
    g1[1] = (int)((nelem & 0xFFFFu) << 16);       // [47:32]=0 | tensor_dim0 lo16
    g1[2] = (int)(((nelem >> 16) & 0xFFFFu)) | (1 << 16); // dim0 hi | tensor_dim1=1 lo
    g1[3] = (int)((nelem & 0xFFFFu) << 16);       // tensor_dim1 hi=0 | tile_dim0
    g1[4] = 1;                                    // tile_dim1=1, tile_dim2=0
    g1[5] = (int)nelem;                           // tensor_dim0_stride[31:0]
    g1[6] = 0;                                    // stride0 hi | stride1 lo
    g1[7] = 0;
    v4i g2 = {0, 0, 0, 0};                        // 2D tensor: groups 2/3 unused
    v4i g3 = {0, 0, 0, 0};
    v8i g4 = {0, 0, 0, 0, 0, 0, 0, 0};
    __builtin_amdgcn_tensor_load_to_lds(g0, g1, g2, g3, g4, /*cpol=*/0);
}

// Problem constants
#define NGRAPH 1024
#define LEADS 12
#define NNODES (NGRAPH*LEADS)      // 12288
#define SIGLEN 2048
#define L1 1024                    // conv1 out length
#define L2 512                     // conv2 out length
#define C1C 16
#define C2C 32
#define K2PAD 96                   // 16*5=80 padded to 3x32
#define CNN_DIM 64
#define HID 64
#define HEADS 4
#define NCLS 9

// ws offsets (bytes, 256-aligned)
#define OFF_C2WH   0u                      // 32*96 f16        = 6144
#define OFF_W1H    6144u                   // 64*256 f16       = 32768
#define OFF_W2H    38912u                  // 256*64 f16       = 32768
#define OFF_XEH    71680u                  // 12288*64 f16     = 1572864
#define OFF_H1     1644544u                // 12288*256 f32    = 12582912
#define OFF_ES1    14227456u               // 12288*4 f32      = 196608
#define OFF_ED1    14424064u               // 12288*4 f32      = 196608
#define OFF_X1H    14620672u               // 12288*256 f16    = 6291456
#define OFF_H2     20912128u               // 12288*64 f32     = 3145728

// ---------------- Kernel 0: weight prep (f32 -> f16 packs) ----------------
__global__ __launch_bounds__(256) void k_prep(
    const float* __restrict__ conv2_w,   // [32,16,5]
    const float* __restrict__ W1,        // [64,256]
    const float* __restrict__ W2,        // [256,64]
    _Float16* __restrict__ c2wh,         // [32,96] zero padded
    _Float16* __restrict__ W1h,          // [64,256]
    _Float16* __restrict__ W2h)          // [256,64]
{
    int i = blockIdx.x * 256 + threadIdx.x;      // 35840 total
    if (i < C2C * K2PAD) {
        int oc = i / K2PAD, kk = i % K2PAD;
        float v = (kk < 80) ? conv2_w[oc * 80 + kk] : 0.f;   // kk = ic*5+k
        c2wh[i] = (_Float16)v;
    } else if (i < C2C * K2PAD + 64 * 256) {
        int j = i - C2C * K2PAD;
        W1h[j] = (_Float16)W1[j];
    } else if (i < C2C * K2PAD + 64 * 256 + 256 * 64) {
        int j = i - C2C * K2PAD - 64 * 256;
        W2h[j] = (_Float16)W2[j];
    }
}

// ---------------- Kernel 1: fused CNN per signal ----------------
// One block = one signal. TDM async-loads the signal into LDS; conv1 (VALU)
// -> LDS f16; conv2 as WMMA implicit GEMM with ReLU+mean epilogue; 32->64
// linear. Output xe f16 [N,64].
__global__ __launch_bounds__(256) void k_cnn(
    const float* __restrict__ x_raw,
    const float* __restrict__ c1w, const float* __restrict__ c1b,
    const float* __restrict__ c2b,
    const _Float16* __restrict__ c2wh,
    const float* __restrict__ lin_w, const float* __restrict__ lin_b,
    _Float16* __restrict__ xe_h)
{
    __shared__ float xs[SIGLEN];
    __shared__ _Float16 c1s[C1C * (L1 + 4)];   // pad 2 each side for conv2 k=5,p=2
    __shared__ float accs[C2C];

    const int tid  = threadIdx.x;
    const int lane = tid & 31;
    const int wave = tid >> 5;
    const int n    = blockIdx.x;

    // wave 0: kick off the TDM DMA of this block's signal (8KB) into LDS
    if (tid < 32) {
        tdm_load_1d_f32((unsigned int)(uintptr_t)&xs[0],
                        x_raw + (size_t)n * SIGLEN, SIGLEN);
    }
    // meanwhile everybody zeros accumulators / pads
    if (tid < C2C) accs[tid] = 0.f;
    if (tid < 64) {
        int ic = tid >> 2, w = tid & 3;
        int idx = (w < 2) ? w : (L1 + w);
        c1s[ic * (L1 + 4) + idx] = (_Float16)0.f;
    }
    if (tid < 32) __builtin_amdgcn_s_wait_tensorcnt(0);   // wave-uniform branch
    __syncthreads();

    // conv1: 16 ch x 1024 pos, k=7 s=2 p=3, ReLU -> f16 LDS
    for (int i = tid; i < C1C * L1; i += 256) {
        int oc = i >> 10, p = i & 1023;
        float s = c1b[oc];
        #pragma unroll
        for (int k = 0; k < 7; ++k) {
            int xi = 2 * p - 3 + k;
            float xv = (xi >= 0 && xi < SIGLEN) ? xs[xi] : 0.f;
            s += c1w[oc * 7 + k] * xv;
        }
        c1s[oc * (L1 + 4) + 2 + p] = (_Float16)fmaxf(s, 0.f);
    }
    __syncthreads();

    // conv2 as WMMA GEMM: [32 x 96pad] x [96pad x 512], fold ReLU+sum epilogue.
    Frag16 a[2][3];
    {
        const int row_in = lane & 15;
        const int khi = (lane >= 16) ? 8 : 0;
        #pragma unroll
        for (int m = 0; m < 2; ++m)
            #pragma unroll
            for (int t = 0; t < 3; ++t) {
                int row = m * 16 + row_in;
                #pragma unroll
                for (int e = 0; e < 16; ++e) {
                    int v = e >> 1, j = e & 1;
                    int kk = 2 * v + 2 * khi + j + 32 * t;
                    a[m][t].h[e] = c2wh[row * K2PAD + kk];
                }
            }
    }
    float s0[8], s1[8];
    #pragma unroll
    for (int r = 0; r < 8; ++r) { s0[r] = 0.f; s1[r] = 0.f; }
    const int col_in = lane & 15;
    const int kb = (lane >= 16) ? 16 : 0;
    const int mo = (lane >= 16) ? 8 : 0;

    for (int it = 0; it < 4; ++it) {
        const int nt = wave + it * 8;          // 8 waves x 4 = 32 N-tiles = 512 pos
        const int p = nt * 16 + col_in;
        v8f c0 = {}; v8f c1a = {};
        #pragma unroll
        for (int t = 0; t < 3; ++t) {
            Frag16 b;
            #pragma unroll
            for (int e = 0; e < 16; ++e) {
                int v = e >> 1, j = e & 1;
                int kk = 2 * v + kb + j + 32 * t;
                _Float16 val = (_Float16)0.f;
                if (kk < 80) {
                    int ic = kk / 5, k = kk - ic * 5;
                    val = c1s[ic * (L1 + 4) + 2 * p + k];   // 2p-2+k with +2 pad
                }
                b.h[e] = val;
            }
            c0  = wmma_f16(a[0][t].v, b.v, c0);
            c1a = wmma_f16(a[1][t].v, b.v, c1a);
        }
        #pragma unroll
        for (int r = 0; r < 8; ++r) {
            s0[r] += fmaxf(c0[r]  + c2b[r + mo],      0.f);
            s1[r] += fmaxf(c1a[r] + c2b[16 + r + mo], 0.f);
        }
    }
    #pragma unroll
    for (int r = 0; r < 8; ++r) {
        atomicAdd(&accs[r + mo],      s0[r]);
        atomicAdd(&accs[16 + r + mo], s1[r]);
    }
    __syncthreads();

    // mean over 512 positions + Linear 32->64
    if (tid < CNN_DIM) {
        float s = lin_b[tid];
        #pragma unroll
        for (int c = 0; c < C2C; ++c)
            s += (accs[c] * (1.f / 512.f)) * lin_w[c * CNN_DIM + tid];
        xe_h[(size_t)n * CNN_DIM + tid] = (_Float16)s;
    }
}

// ---------------- Generic WMMA GEMM: C(f32,MxN) = A(f16,MxK) * B(f16,KxN) ----------------
// Row-major A/B/C; M,N multiples of 16; K multiple of 32. One 16x16 tile per wave.
__global__ __launch_bounds__(256) void k_gemm(
    const _Float16* __restrict__ A, const _Float16* __restrict__ B,
    float* __restrict__ C, int M, int N, int K)
{
    const int lane = threadIdx.x & 31;
    const int wid = blockIdx.x * (blockDim.x >> 5) + (threadIdx.x >> 5);
    const int ntiles = N >> 4;
    const int mt = wid / ntiles, nt = wid % ntiles;
    if (mt * 16 >= M) return;   // wave-uniform

    const int row = mt * 16 + (lane & 15);
    const int col = nt * 16 + (lane & 15);
    const int khiA = (lane >= 16) ? 8 : 0;
    const int kbB  = (lane >= 16) ? 16 : 0;

    v8f acc = {};
    for (int kt = 0; kt < K; kt += 32) {
        Frag16 a, b;
        #pragma unroll
        for (int e = 0; e < 16; ++e) {
            int v = e >> 1, j = e & 1;
            int kka = 2 * v + 2 * khiA + j;
            a.h[e] = A[(size_t)row * K + kt + kka];
            int kkb = 2 * v + kbB + j;
            b.h[e] = B[(size_t)(kt + kkb) * N + col];
        }
        acc = wmma_f16(a.v, b.v, acc);
    }
    const int mo = (lane >= 16) ? 8 : 0;
    #pragma unroll
    for (int r = 0; r < 8; ++r)
        C[(size_t)(mt * 16 + r + mo) * N + nt * 16 + (lane & 15)] = acc[r];
}

// ---------------- GAT1 attention scores ----------------
__global__ __launch_bounds__(256) void k_att1(
    const float* __restrict__ H1, const float* __restrict__ asrc,
    const float* __restrict__ adst, float* __restrict__ es, float* __restrict__ ed)
{
    int i = blockIdx.x * 256 + threadIdx.x;      // NNODES*HEADS = 49152
    if (i >= NNODES * HEADS) return;
    int node = i >> 2, h = i & 3;
    const float* hp = H1 + (size_t)node * (HEADS * HID) + h * HID;
    float s = 0.f, d = 0.f;
    #pragma unroll 4
    for (int k = 0; k < HID; ++k) {
        float v = hp[k];
        s += v * asrc[h * HID + k];
        d += v * adst[h * HID + k];
    }
    es[i] = s; ed[i] = d;
}

// ---------------- GAT1: per-graph softmax + aggregate + bias + ReLU ----------------
__global__ __launch_bounds__(256) void k_gat1(
    const float* __restrict__ H1, const float* __restrict__ es,
    const float* __restrict__ ed, const float* __restrict__ bias1,
    _Float16* __restrict__ x1h)
{
    __shared__ float h1s[LEADS * HEADS * HID];   // 12x256
    __shared__ float esL[LEADS * HEADS], edL[LEADS * HEADS];
    __shared__ float al[LEADS * LEADS * HEADS];  // [s][d][h]
    const int g = blockIdx.x, tid = threadIdx.x;
    const int base = g * LEADS;

    for (int i = tid; i < LEADS * 256; i += 256) h1s[i] = H1[(size_t)base * 256 + i];
    if (tid < LEADS * HEADS) { esL[tid] = es[base * HEADS + tid]; edL[tid] = ed[base * HEADS + tid]; }
    __syncthreads();

    for (int i = tid; i < LEADS * LEADS * HEADS; i += 256) {
        int s = i / (LEADS * HEADS);
        int rem = i - s * (LEADS * HEADS);
        int d = rem >> 2, h = rem & 3;
        float e = esL[s * HEADS + h] + edL[d * HEADS + h];
        al[i] = (e > 0.f) ? e : 0.2f * e;        // leaky_relu(0.2)
    }
    __syncthreads();

    if (tid < LEADS * HEADS) {                   // per (dst, head) softmax over 12 srcs
        float m = -1e30f;
        for (int s = 0; s < LEADS; ++s) m = fmaxf(m, al[s * (LEADS * HEADS) + tid]);
        float Z = 0.f;
        for (int s = 0; s < LEADS; ++s) {
            float ex = expf(al[s * (LEADS * HEADS) + tid] - m);
            al[s * (LEADS * HEADS) + tid] = ex; Z += ex;
        }
        float inv = 1.f / (Z + 1e-16f);
        for (int s = 0; s < LEADS; ++s) al[s * (LEADS * HEADS) + tid] *= inv;
    }
    __syncthreads();

    for (int i = tid; i < LEADS * 256; i += 256) {
        int d = i >> 8, hd = i & 255, h = hd >> 6;
        float s = bias1[hd];
        #pragma unroll
        for (int sn = 0; sn < LEADS; ++sn)
            s += al[sn * (LEADS * HEADS) + d * HEADS + h] * h1s[sn * 256 + hd];
        x1h[(size_t)(base + d) * 256 + hd] = (_Float16)fmaxf(s, 0.f);
    }
}

// ---------------- GAT2 + mean-pool + FC, fully fused per graph ----------------
__global__ __launch_bounds__(128) void k_gat2(
    const float* __restrict__ H2, const float* __restrict__ asrc2,
    const float* __restrict__ adst2, const float* __restrict__ bias2,
    const float* __restrict__ fc_w, const float* __restrict__ fc_b,
    float* __restrict__ out)
{
    __shared__ float h2s[LEADS * HID];
    __shared__ float es2[LEADS], ed2[LEADS];
    __shared__ float al2[LEADS * LEADS];
    __shared__ float x2[LEADS * HID];
    __shared__ float pooled[HID];
    const int g = blockIdx.x, tid = threadIdx.x;

    for (int i = tid; i < LEADS * HID; i += 128) h2s[i] = H2[(size_t)g * LEADS * HID + i];
    __syncthreads();
    if (tid < LEADS) {
        float s = 0.f, d = 0.f;
        #pragma unroll 4
        for (int k = 0; k < HID; ++k) {
            float v = h2s[tid * HID + k];
            s += v * asrc2[k]; d += v * adst2[k];
        }
        es2[tid] = s; ed2[tid] = d;
    }
    __syncthreads();
    for (int i = tid; i < LEADS * LEADS; i += 128) {
        int s = i / LEADS, d = i - s * LEADS;
        float e = es2[s] + ed2[d];
        al2[i] = (e > 0.f) ? e : 0.2f * e;
    }
    __syncthreads();
    if (tid < LEADS) {                           // per-dst softmax
        int d = tid;
        float m = -1e30f;
        for (int s = 0; s < LEADS; ++s) m = fmaxf(m, al2[s * LEADS + d]);
        float Z = 0.f;
        for (int s = 0; s < LEADS; ++s) { float ex = expf(al2[s * LEADS + d] - m); al2[s * LEADS + d] = ex; Z += ex; }
        float inv = 1.f / (Z + 1e-16f);
        for (int s = 0; s < LEADS; ++s) al2[s * LEADS + d] *= inv;
    }
    __syncthreads();
    for (int i = tid; i < LEADS * HID; i += 128) {
        int d = i >> 6, k = i & 63;
        float s = bias2[k];
        #pragma unroll
        for (int sn = 0; sn < LEADS; ++sn) s += al2[sn * LEADS + d] * h2s[sn * HID + k];
        x2[i] = fmaxf(s, 0.f);
    }
    __syncthreads();
    if (tid < HID) {
        float s = 0.f;
        #pragma unroll
        for (int d = 0; d < LEADS; ++d) s += x2[d * HID + tid];
        pooled[tid] = s * (1.f / (float)LEADS);
    }
    __syncthreads();
    if (tid < NCLS) {
        float s = fc_b[tid];
        #pragma unroll 8
        for (int k = 0; k < HID; ++k) s += pooled[k] * fc_w[k * NCLS + tid];
        out[g * NCLS + tid] = s;
    }
}

// ---------------- launch ----------------
extern "C" void kernel_launch(void* const* d_in, const int* in_sizes, int n_in,
                              void* d_out, int out_size, void* d_ws, size_t ws_size,
                              hipStream_t stream) {
    const float* x_raw    = (const float*)d_in[0];
    // d_in[1] edge_index, d_in[2] batch: fixed complete-graph structure, unused
    const float* conv1_w  = (const float*)d_in[3];
    const float* conv1_b  = (const float*)d_in[4];
    const float* conv2_w  = (const float*)d_in[5];
    const float* conv2_b  = (const float*)d_in[6];
    const float* lin_w    = (const float*)d_in[7];
    const float* lin_b    = (const float*)d_in[8];
    const float* W1       = (const float*)d_in[9];
    const float* att_src1 = (const float*)d_in[10];
    const float* att_dst1 = (const float*)d_in[11];
    const float* bias1    = (const float*)d_in[12];
    const float* W2       = (const float*)d_in[13];
    const float* att_src2 = (const float*)d_in[14];
    const float* att_dst2 = (const float*)d_in[15];
    const float* bias2    = (const float*)d_in[16];
    const float* fc_w     = (const float*)d_in[17];
    const float* fc_b     = (const float*)d_in[18];

    char* ws = (char*)d_ws;
    _Float16* c2wh = (_Float16*)(ws + OFF_C2WH);
    _Float16* W1h  = (_Float16*)(ws + OFF_W1H);
    _Float16* W2h  = (_Float16*)(ws + OFF_W2H);
    _Float16* xe_h = (_Float16*)(ws + OFF_XEH);
    float*    H1   = (float*)(ws + OFF_H1);
    float*    es1  = (float*)(ws + OFF_ES1);
    float*    ed1  = (float*)(ws + OFF_ED1);
    _Float16* x1h  = (_Float16*)(ws + OFF_X1H);
    float*    H2   = (float*)(ws + OFF_H2);

    // 0) weight packs (35840 elems)
    k_prep<<<140, 256, 0, stream>>>(conv2_w, W1, W2, c2wh, W1h, W2h);
    // 1) fused CNN: TDM load -> conv1 -> conv2(WMMA) -> ReLU+mean -> linear
    k_cnn<<<NNODES, 256, 0, stream>>>(x_raw, conv1_w, conv1_b, conv2_b,
                                      c2wh, lin_w, lin_b, xe_h);
    // 2) H1 = xe @ W1  (12288x256, K=64): 768*16 = 12288 wave-tiles / 8 = 1536 blocks
    k_gemm<<<1536, 256, 0, stream>>>(xe_h, W1h, H1, NNODES, HEADS * HID, CNN_DIM);
    // 3) attention scores for GAT1
    k_att1<<<(NNODES * HEADS) / 256, 256, 0, stream>>>(H1, att_src1, att_dst1, es1, ed1);
    // 4) GAT1 softmax + aggregate + bias + ReLU -> x1 f16
    k_gat1<<<NGRAPH, 256, 0, stream>>>(H1, es1, ed1, bias1, x1h);
    // 5) H2 = x1 @ W2  (12288x64, K=256): 768*4 = 3072 wave-tiles / 8 = 384 blocks
    k_gemm<<<384, 256, 0, stream>>>(x1h, W2h, H2, NNODES, HID, HEADS * HID);
    // 6) GAT2 + mean-pool + FC -> out [1024,9]
    k_gat2<<<NGRAPH, 128, 0, stream>>>(H2, att_src2, att_dst2, bias2,
                                       fc_w, fc_b, (float*)d_out);
}
